// DeepLSTMCell_15977278341628
// MI455X (gfx1250) — matrix operand order repaired and checked by reading, not compiled, not run on hardware
//
#include <hip/hip_runtime.h>

typedef __attribute__((ext_vector_type(16))) _Float16 v16h;
typedef __attribute__((ext_vector_type(8)))  float    v8f;

#define B_    64
#define T_    512
#define DIN_  512
#define H_    1024
#define NBLK  32
#define NTHR  256
#define FRAGBUF_ (4 * 32 * 512)   // f16 per h-buffer parity: [mT=4][kc=32][lane=32][16]

__device__ __forceinline__ float fsig(float x)  { return 1.0f / (1.0f + __expf(-x)); }
__device__ __forceinline__ float ftanh(float x) { return 2.0f * fsig(2.0f * x) - 1.0f; }

// ---------------- init: barriers, h buffers, fused biases ----------------
__global__ void k_init(unsigned* __restrict__ bar, unsigned* __restrict__ h0z,
                       unsigned* __restrict__ h1z,
                       float* __restrict__ bias0, float* __restrict__ bias1,
                       const float* __restrict__ bih0, const float* __restrict__ bhh0,
                       const float* __restrict__ bih1, const float* __restrict__ bhh1) {
  int i = blockIdx.x * blockDim.x + threadIdx.x;
  if (i < 1024) bar[i] = 0u;
  if (i < 65536) { h0z[i] = 0u; h1z[i] = 0u; }   // 2 parities x 64Kf16 each (u32 writes)
  if (i < 4096) { bias0[i] = bih0[i] + bhh0[i]; bias1[i] = bih1[i] + bhh1[i]; }
}

// ---------------- pack weight W[4096,K] into B-fragments (16x16x32 f16) ----
// B layout: lane 0-15 -> col n, VGPR d: K = kc*32 + (d>>2)*16 + half*8 + (d&3)*2.
__global__ void k_packW(const float* __restrict__ W, int K, int kChunks,
                        _Float16* __restrict__ dst) {
  int idx = blockIdx.x * blockDim.x + threadIdx.x;
  int total = 256 * kChunks * 32 * 8;
  if (idx >= total) return;
  int d    = idx & 7;
  int lane = (idx >> 3) & 31;
  int f    = idx >> 8;            // nt*kChunks + kc
  int kc   = f % kChunks;
  int nt   = f / kChunks;
  int n = lane & 15, half = lane >> 4;
  int k0 = kc * 32 + (d >> 2) * 16 + half * 8 + (d & 3) * 2;
  int j  = nt * 16 + n;
  const float* wr = W + (size_t)j * K + k0;
  _Float16* p = dst + ((size_t)f * 32 + lane) * 16 + d * 2;
  p[0] = (_Float16)wr[0];
  p[1] = (_Float16)wr[1];
}

// ---------------- pack x[B,T,DIN] into A-fragments ----------------
// A layout: lane 0-15 -> row m, VGPR d: K = kc*32 + (d>>2)*16 + half*8 + (d&3)*2.
__global__ void k_packX(const float* __restrict__ x, _Float16* __restrict__ dst) {
  int idx  = blockIdx.x * blockDim.x + threadIdx.x;   // < 512*4*16*256
  int d    = idx & 7;
  int lane = (idx >> 3) & 31;
  int f    = idx >> 8;            // (t*4+mT)*16 + kc
  int kc   = f & 15;
  int mT   = (f >> 4) & 3;
  int t    = f >> 6;
  if (t >= T_) return;
  int m = lane & 15, half = lane >> 4;
  int b  = mT * 16 + m;
  int k0 = kc * 32 + (d >> 2) * 16 + half * 8 + (d & 3) * 2;
  const float* xr = x + ((size_t)b * T_ + t) * DIN_ + k0;
  _Float16* p = dst + ((size_t)f * 32 + lane) * 16 + d * 2;
  p[0] = (_Float16)xr[0];
  p[1] = (_Float16)xr[1];
}

// slot-per-event grid barrier (slots pre-zeroed each call -> deterministic)
__device__ __forceinline__ void gridBarrier(unsigned* __restrict__ bar, int slot) {
  __syncthreads();
  if (threadIdx.x == 0) {
    __threadfence();
    atomicAdd(&bar[slot], 1u);
    while (__hip_atomic_load(&bar[slot], __ATOMIC_ACQUIRE,
                             __HIP_MEMORY_SCOPE_AGENT) < (unsigned)NBLK) {
      __builtin_amdgcn_s_sleep(1);
    }
  }
  __syncthreads();
  __threadfence();
}

// ---------------- persistent 2-layer LSTM ----------------
// 32 blocks x 8 waves = 256 waves; wave -> (mT 0..3, nHT 0..63) owns all 4 gate
// tiles of one (batch-tile, hidden-tile) patch; c0/c1 live in VGPRs for all T.
__global__ __launch_bounds__(NTHR, 1)
void k_lstm(const _Float16* __restrict__ xp,
            const _Float16* __restrict__ wih0, const _Float16* __restrict__ whh0,
            const _Float16* __restrict__ wih1, const _Float16* __restrict__ whh1,
            const float* __restrict__ bias0, const float* __restrict__ bias1,
            _Float16* __restrict__ h0b, _Float16* __restrict__ h1b,
            unsigned* __restrict__ bar, float* __restrict__ out) {
  const int lane = threadIdx.x & 31;
  const int wave = blockIdx.x * (NTHR / 32) + (threadIdx.x >> 5);  // 0..255
  const int mT   = wave >> 6;
  const int nHT  = wave & 63;
  const int half = lane >> 4;
  const int nCol = lane & 15;
  const int col  = nHT * 16 + nCol;          // hidden column of this D tile

  // ---- precompute the A-fragment scatter position for writing h (k = col) --
  const int w_kc    = col >> 5;
  const int w_ko    = col & 31;
  const int w_group = w_ko >> 4;
  const int w_hk    = w_ko & 15;
  const int w_halfA = w_hk >> 3;
  const int w_d     = w_group * 4 + ((w_hk & 7) >> 1);
  const int w_e     = 2 * w_d + (col & 1);
  // element (m,col) lives at frag (mT*32+w_kc), laneA = (m&15)+16*w_halfA, slot w_e
  const size_t w_base =
      ((size_t)(mT * 32 + w_kc) * 32 + (size_t)w_halfA * 16) * 16 + w_e;

  float b0v[4], b1v[4];
#pragma unroll
  for (int g = 0; g < 4; ++g) {
    b0v[g] = bias0[g * 1024 + col];
    b1v[g] = bias1[g * 1024 + col];
  }

  v8f c0 = {}, c1 = {};

  for (int t = 0; t < T_; ++t) {
    const int wr = t & 1, rd = wr ^ 1;

    // ================= layer 0: gates = x_t@Wih0^T + h0@Whh0^T + b =========
    v8f acc[4];
#pragma unroll
    for (int g = 0; g < 4; ++g) {
      v8f a;
#pragma unroll
      for (int e = 0; e < 8; ++e) a[e] = b0v[g];
      acc[g] = a;
    }
    const _Float16* xfrag = xp + ((size_t)(t * 4 + mT) * 16) * 512 + lane * 16;
#pragma unroll 2
    for (int kc = 0; kc < 16; ++kc) {
      v16h a = *(const v16h*)(xfrag + (size_t)kc * 512);
      __builtin_prefetch(xfrag + (size_t)(kc + 2) * 512, 0, 3);
#pragma unroll
      for (int g = 0; g < 4; ++g) {
        const _Float16* bp =
            wih0 + ((size_t)((g * 64 + nHT) * 16 + kc) * 512) + lane * 16;
        v16h b = *(const v16h*)bp;
        acc[g] = __builtin_amdgcn_wmma_f32_16x16x32_f16(false, a, false, b,
                                                        (short)0, acc[g], false, false);
      }
    }
    const _Float16* h0rd =
        h0b + (size_t)rd * FRAGBUF_ + (size_t)mT * 32 * 512 + (size_t)lane * 16;
#pragma unroll 2
    for (int kc = 0; kc < 32; ++kc) {
      v16h a = *(const v16h*)(h0rd + (size_t)kc * 512);
#pragma unroll
      for (int g = 0; g < 4; ++g) {
        const _Float16* bp =
            whh0 + ((size_t)((g * 64 + nHT) * 32 + kc) * 512) + lane * 16;
        v16h b = *(const v16h*)bp;
        acc[g] = __builtin_amdgcn_wmma_f32_16x16x32_f16(false, a, false, b,
                                                        (short)0, acc[g], false, false);
      }
    }
    // i,f,g,o local to this wave -> update c0 in registers, emit h0 fragments
    _Float16* h0wr = h0b + (size_t)wr * FRAGBUF_ + w_base;
#pragma unroll
    for (int mi = 0; mi < 8; ++mi) {
      float iv = fsig(acc[0][mi]);
      float fv = fsig(acc[1][mi]);
      float gv = ftanh(acc[2][mi]);
      float ov = fsig(acc[3][mi]);
      float cn = fv * c0[mi] + iv * gv;
      c0[mi] = cn;
      float hv = ov * ftanh(cn);
      int laneA = half * 8 + mi;              // (m & 15)
      h0wr[(size_t)laneA * 16] = (_Float16)hv;
    }
    __threadfence();
    gridBarrier(bar, 2 * t);

    // ================= layer 1: gates = h0_t@Wih1^T + h1@Whh1^T + b =========
#pragma unroll
    for (int g = 0; g < 4; ++g) {
      v8f a;
#pragma unroll
      for (int e = 0; e < 8; ++e) a[e] = b1v[g];
      acc[g] = a;
    }
    const _Float16* h0now =
        h0b + (size_t)wr * FRAGBUF_ + (size_t)mT * 32 * 512 + (size_t)lane * 16;
    const _Float16* h1rd =
        h1b + (size_t)rd * FRAGBUF_ + (size_t)mT * 32 * 512 + (size_t)lane * 16;
#pragma unroll 2
    for (int kc = 0; kc < 32; ++kc) {
      v16h a = *(const v16h*)(h0now + (size_t)kc * 512);
#pragma unroll
      for (int g = 0; g < 4; ++g) {
        const _Float16* bp =
            wih1 + ((size_t)((g * 64 + nHT) * 32 + kc) * 512) + lane * 16;
        v16h b = *(const v16h*)bp;
        acc[g] = __builtin_amdgcn_wmma_f32_16x16x32_f16(false, a, false, b,
                                                        (short)0, acc[g], false, false);
      }
    }
#pragma unroll 2
    for (int kc = 0; kc < 32; ++kc) {
      v16h a = *(const v16h*)(h1rd + (size_t)kc * 512);
#pragma unroll
      for (int g = 0; g < 4; ++g) {
        const _Float16* bp =
            whh1 + ((size_t)((g * 64 + nHT) * 32 + kc) * 512) + lane * 16;
        v16h b = *(const v16h*)bp;
        acc[g] = __builtin_amdgcn_wmma_f32_16x16x32_f16(false, a, false, b,
                                                        (short)0, acc[g], false, false);
      }
    }
    _Float16* h1wr = h1b + (size_t)wr * FRAGBUF_ + w_base;
#pragma unroll
    for (int mi = 0; mi < 8; ++mi) {
      float iv = fsig(acc[0][mi]);
      float fv = fsig(acc[1][mi]);
      float gv = ftanh(acc[2][mi]);
      float ov = fsig(acc[3][mi]);
      float cn = fv * c1[mi] + iv * gv;
      c1[mi] = cn;
      float hv = ov * ftanh(cn);
      int laneA = half * 8 + mi;
      h1wr[(size_t)laneA * 16] = (_Float16)hv;
      int m = mT * 16 + half * 8 + mi;
      out[((size_t)m * T_ + t) * H_ + col] = fsig(hv);   // final sigmoid(h1)
    }
    __threadfence();
    gridBarrier(bar, 2 * t + 1);
  }
}

extern "C" void kernel_launch(void* const* d_in, const int* in_sizes, int n_in,
                              void* d_out, int out_size, void* d_ws, size_t ws_size,
                              hipStream_t stream) {
  const float* x    = (const float*)d_in[0];
  const float* Wih0 = (const float*)d_in[1];
  const float* Whh0 = (const float*)d_in[2];
  const float* bih0 = (const float*)d_in[3];
  const float* bhh0 = (const float*)d_in[4];
  const float* Wih1 = (const float*)d_in[5];
  const float* Whh1 = (const float*)d_in[6];
  const float* bih1 = (const float*)d_in[7];
  const float* bhh1 = (const float*)d_in[8];
  float* out = (float*)d_out;
  char* ws = (char*)d_ws;

  constexpr size_t OFF_BAR  = 0;                       // 1024 u32
  constexpr size_t OFF_B0   = 4096;                    // 4096 f32
  constexpr size_t OFF_B1   = OFF_B0 + 4096 * 4;
  constexpr size_t OFF_H0   = OFF_B1 + 4096 * 4;       // 2 x FRAGBUF_ f16
  constexpr size_t OFF_H1   = OFF_H0 + 2 * (size_t)FRAGBUF_ * 2;
  constexpr size_t OFF_WIH0 = OFF_H1 + 2 * (size_t)FRAGBUF_ * 2;
  constexpr size_t SZ_WIH0  = (size_t)256 * 16 * 512 * 2;   // 4 MiB
  constexpr size_t OFF_WHH0 = OFF_WIH0 + SZ_WIH0;
  constexpr size_t SZ_WHH   = (size_t)256 * 32 * 512 * 2;   // 8 MiB
  constexpr size_t OFF_WIH1 = OFF_WHH0 + SZ_WHH;
  constexpr size_t OFF_WHH1 = OFF_WIH1 + SZ_WHH;
  constexpr size_t OFF_XP   = OFF_WHH1 + SZ_WHH;
  constexpr size_t SZ_XP    = (size_t)512 * 4 * 16 * 512 * 2;  // 32 MiB

  if (ws_size < OFF_XP + SZ_XP) return;

  unsigned* bar   = (unsigned*)(ws + OFF_BAR);
  float* bias0    = (float*)(ws + OFF_B0);
  float* bias1    = (float*)(ws + OFF_B1);
  _Float16* h0b   = (_Float16*)(ws + OFF_H0);
  _Float16* h1b   = (_Float16*)(ws + OFF_H1);
  _Float16* wih0p = (_Float16*)(ws + OFF_WIH0);
  _Float16* whh0p = (_Float16*)(ws + OFF_WHH0);
  _Float16* wih1p = (_Float16*)(ws + OFF_WIH1);
  _Float16* whh1p = (_Float16*)(ws + OFF_WHH1);
  _Float16* xp    = (_Float16*)(ws + OFF_XP);

  k_init<<<(65536 + 255) / 256, 256, 0, stream>>>(bar, (unsigned*)h0b, (unsigned*)h1b,
                                                  bias0, bias1, bih0, bhh0, bih1, bhh1);
  k_packW<<<(256 * 16 * 256) / 256, 256, 0, stream>>>(Wih0, 512, 16, wih0p);
  k_packW<<<(256 * 32 * 256) / 256, 256, 0, stream>>>(Whh0, 1024, 32, whh0p);
  k_packW<<<(256 * 32 * 256) / 256, 256, 0, stream>>>(Wih1, 1024, 32, wih1p);
  k_packW<<<(256 * 32 * 256) / 256, 256, 0, stream>>>(Whh1, 1024, 32, whh1p);
  k_packX<<<(512 * 4 * 16 * 256) / 256, 256, 0, stream>>>(x, xp);
  k_lstm<<<NBLK, NTHR, 0, stream>>>(xp, wih0p, whh0p, wih1p, whh1p,
                                    bias0, bias1, h0b, h1b, bar, out);
}
